// Attention_baseline_64003602645394
// MI455X (gfx1250) — compile-verified, matrix operand order checked by reading, not run on hardware
//
#include <hip/hip_runtime.h>
#include <hip/hip_bf16.h>
#include <math.h>

// Problem dims (fixed by the reference)
#define BB 64
#define SS 512
#define DD 1024
#define EE 1024
#define DW 2048   // D + E2, row stride of w_weight

typedef __attribute__((ext_vector_type(16))) __bf16 v16bf;
typedef __attribute__((ext_vector_type(8)))  __bf16 v8bf;
typedef __attribute__((ext_vector_type(8)))  float  v8f;

union V16 { v16bf v; v8bf h[2]; };

// Branchless tanh: hardware V_TANH_F32 on gfx1250 if the builtin exists,
// otherwise a 5-op exp-based formulation (saturates correctly at +/-1).
__device__ __forceinline__ float fast_tanhf(float x) {
#if defined(__has_builtin)
#if __has_builtin(__builtin_amdgcn_tanhf)
    return __builtin_amdgcn_tanhf(x);
#else
    const float e = __expf(2.0f * x);
    return 1.0f - 2.0f / (e + 1.0f);
#endif
#else
    const float e = __expf(2.0f * x);
    return 1.0f - 2.0f / (e + 1.0f);
#endif
}

// ---------------------------------------------------------------------------
// K0: pack We (= w_weight[:, D:]) from strided fp32 into contiguous bf16 [D,E]
// ---------------------------------------------------------------------------
__global__ void cvt_we_kernel(const float* __restrict__ W, __bf16* __restrict__ weB)
{
    int idx4 = (blockIdx.x * 256 + threadIdx.x) * 4;
    int k = idx4 >> 10;
    int e = idx4 & 1023;
    const float4 f = *reinterpret_cast<const float4*>(W + (size_t)k * DW + DD + e);
    __bf16* d = weB + idx4;
    d[0] = (__bf16)f.x; d[1] = (__bf16)f.y; d[2] = (__bf16)f.z; d[3] = (__bf16)f.w;
}

// ---------------------------------------------------------------------------
// K1: dec_proj[b,k] = sum_d dec[b,d] * w_weight[k,d] + bias[k]   (64x1024)
// ---------------------------------------------------------------------------
__global__ void decproj_kernel(const float* __restrict__ dec,
                               const float* __restrict__ W,
                               const float* __restrict__ bias,
                               float* __restrict__ dp)
{
    const int k = blockIdx.x;
    const int t = threadIdx.x;                // 0..63 == batch index
    __shared__ float row[DD];
    for (int i = t; i < DD; i += 64) row[i] = W[(size_t)k * DW + i];
    __syncthreads();
    const float* dv = dec + (size_t)t * DD;
    float acc = 0.f;
#pragma unroll 8
    for (int d = 0; d < DD; ++d) acc = fmaf(dv[d], row[d], acc);
    dp[(size_t)t * DD + k] = acc + bias[k];
}

// ---------------------------------------------------------------------------
// K2: fused energy GEMM + tanh + dot(v) -> logits[b*S+s]
// One wave32 per 32-row tile (two 16-row A tiles in 64 KB LDS).
// Each B-operand load pair feeds TWO v_wmma_f32_16x16x32_bf16 ops,
// halving L2 traffic for We versus one-tile-per-wave.
// 4096 WMMAs per wave.
// ---------------------------------------------------------------------------
__global__ __launch_bounds__(32)
void energy_kernel(const float* __restrict__ enc,      // [B*S, E] fp32
                   const __bf16* __restrict__ weB,     // [D, E] bf16 packed
                   const float* __restrict__ dec_proj, // [B, D] (bias folded in)
                   const float* __restrict__ vw,       // [D]
                   float* __restrict__ logits)         // [B*S]
{
    const int lane    = threadIdx.x;            // 0..31
    const int rowBase = blockIdx.x * 32;        // 32 | S -> tile never crosses b
    const int b       = rowBase / SS;

    __shared__ __bf16 smem[32 * EE];            // 64 KB, private to this wave

    // --- stage 32 rows of enc into LDS as bf16 (each element touched once) ---
    for (int i = 0; i < 256; ++i) {
        const int pos = (i * 32 + lane) * 4;    // 0..32767
        const int row = pos >> 10;
        const int e   = pos & 1023;
        const float4 f = *reinterpret_cast<const float4*>(
            enc + (size_t)(rowBase + row) * EE + e);
        __bf16* d = smem + pos;
        d[0] = (__bf16)f.x; d[1] = (__bf16)f.y; d[2] = (__bf16)f.z; d[3] = (__bf16)f.w;
    }
    // Single wave owns this LDS region: DS ops are in-order per wave, no barrier.

    const int half = lane >> 4;                 // 0: K 0..7/16..23, 1: K 8..15/24..31
    const int l    = lane & 15;                 // A row / B column within tile
    const int eOff = half * 8;

    const __bf16* a0 = smem + (size_t)l * EE;          // rows rowBase+0..15
    const __bf16* a1 = smem + (size_t)(16 + l) * EE;   // rows rowBase+16..31

    v8f att0 = {0.f,0.f,0.f,0.f,0.f,0.f,0.f,0.f};
    v8f att1 = {0.f,0.f,0.f,0.f,0.f,0.f,0.f,0.f};

    for (int kt = 0; kt < 64; ++kt) {
        const int k0 = kt * 16;
        const __bf16* bRow = weB + (size_t)(k0 + l) * EE;
        // hoisted: latency hides under the 64-WMMA chain below
        const float dp = dec_proj[(size_t)b * DD + k0 + l];
        const float vk = vw[k0 + l];

        v8f acc0 = {0.f,0.f,0.f,0.f,0.f,0.f,0.f,0.f};
        v8f acc1 = {0.f,0.f,0.f,0.f,0.f,0.f,0.f,0.f};
#pragma unroll 4
        for (int ec = 0; ec < 32; ++ec) {
            const int e0 = ec * 32;
            V16 Bv, A0, A1;
            // B operand: lane n holds column n = row (k0+n) of We
            Bv.h[0] = *reinterpret_cast<const v8bf*>(bRow + e0 + eOff);
            Bv.h[1] = *reinterpret_cast<const v8bf*>(bRow + e0 + eOff + 16);
            // A operands from LDS (16-bit A layout: half picks K-subgroups)
            A0.h[0] = *reinterpret_cast<const v8bf*>(a0 + e0 + eOff);
            A0.h[1] = *reinterpret_cast<const v8bf*>(a0 + e0 + eOff + 16);
            A1.h[0] = *reinterpret_cast<const v8bf*>(a1 + e0 + eOff);
            A1.h[1] = *reinterpret_cast<const v8bf*>(a1 + e0 + eOff + 16);
            acc0 = __builtin_amdgcn_wmma_f32_16x16x32_bf16(
                false, A0.v, false, Bv.v, (short)0, acc0, false, false);
            acc1 = __builtin_amdgcn_wmma_f32_16x16x32_bf16(
                false, A1.v, false, Bv.v, (short)0, acc1, false, false);
        }
        // energy(m, k0+l): lanes 0-15 -> m=j, lanes 16-31 -> m=8+j
#pragma unroll
        for (int j = 0; j < 8; ++j) {
            att0[j] += fast_tanhf(acc0[j] + dp) * vk;
            att1[j] += fast_tanhf(acc1[j] + dp) * vk;
        }
    }

    // reduce the 16 columns held by each half-wave (masks < 16 stay in-half)
#pragma unroll
    for (int m = 1; m <= 8; m <<= 1) {
#pragma unroll
        for (int j = 0; j < 8; ++j) {
            att0[j] += __shfl_xor(att0[j], m, 32);
            att1[j] += __shfl_xor(att1[j], m, 32);
        }
    }
    if (l == 0) {                               // lane 0 -> rows m=0..7, lane 16 -> m=8..15
        float* dst0 = logits + rowBase + half * 8;
        float* dst1 = logits + rowBase + 16 + half * 8;
#pragma unroll
        for (int j = 0; j < 8; ++j) { dst0[j] = att0[j]; dst1[j] = att1[j]; }
    }
}

// ---------------------------------------------------------------------------
// K3: masked softmax over S per batch row (one block of 512 per b)
// ---------------------------------------------------------------------------
__global__ void softmax_kernel(const float* __restrict__ logits,
                               const int* __restrict__ mask,
                               float* __restrict__ weights)
{
    const int b = blockIdx.x;
    const int t = threadIdx.x;                  // 0..511
    __shared__ float red[SS];
    float x = logits[(size_t)b * SS + t];
    if (mask[(size_t)b * SS + t] == 0) x = -INFINITY;
    red[t] = x; __syncthreads();
    for (int off = 256; off > 0; off >>= 1) {
        if (t < off) red[t] = fmaxf(red[t], red[t + off]);
        __syncthreads();
    }
    const float mx = red[0]; __syncthreads();
    const float e = __expf(x - mx);
    red[t] = e; __syncthreads();
    for (int off = 256; off > 0; off >>= 1) {
        if (t < off) red[t] += red[t + off];
        __syncthreads();
    }
    const float s = red[0];
    weights[(size_t)b * SS + t] = e / s;
}

// ---------------------------------------------------------------------------
// K4: att_out[b,e] = sum_s w[b,s] * enc[b,s,e]  (bandwidth bound, coalesced)
// ---------------------------------------------------------------------------
__global__ void wsum_kernel(const float* __restrict__ enc,
                            const float* __restrict__ weights,
                            float* __restrict__ out)
{
    const int b  = blockIdx.x >> 2;
    const int ec = (blockIdx.x & 3) * 256 + threadIdx.x;
    __shared__ float w[SS];
    w[threadIdx.x]       = weights[(size_t)b * SS + threadIdx.x];
    w[threadIdx.x + 256] = weights[(size_t)b * SS + threadIdx.x + 256];
    __syncthreads();
    const float* ep = enc + (size_t)b * SS * EE + ec;
    float acc = 0.f;
#pragma unroll 8
    for (int s = 0; s < SS; ++s)
        acc = fmaf(w[s], ep[(size_t)s * EE], acc);
    out[(size_t)b * EE + ec] = acc;
}

// ---------------------------------------------------------------------------
extern "C" void kernel_launch(void* const* d_in, const int* in_sizes, int n_in,
                              void* d_out, int out_size, void* d_ws, size_t ws_size,
                              hipStream_t stream) {
    const float* dec   = (const float*)d_in[0];   // [B, D]
    const float* enc   = (const float*)d_in[1];   // [B, S, E]
    const int*   mask  = (const int*)  d_in[2];   // [B, S]
    const float* W     = (const float*)d_in[3];   // [D, D+E]
    const float* bias  = (const float*)d_in[4];   // [D]
    const float* vw    = (const float*)d_in[5];   // [D]
    float*       out   = (float*)d_out;           // [B, 1, E]

    // workspace carve-up (2.5 MB total)
    char* ws = (char*)d_ws;
    __bf16* weB     = (__bf16*)(ws);                               // 2 MB
    float*  decproj = (float*)(ws + 2097152);                      // 256 KB
    float*  logits  = (float*)(ws + 2097152 + 262144);             // 128 KB
    float*  weights = (float*)(ws + 2097152 + 262144 + 131072);    // 128 KB

    // K0: pack We -> bf16
    cvt_we_kernel<<<(DD * EE) / (256 * 4), 256, 0, stream>>>(W, weB);
    // K1: dec_proj + bias
    decproj_kernel<<<DD, 64, 0, stream>>>(dec, W, bias, decproj);
    // K2: fused bf16-WMMA energy GEMM + tanh + v-reduction -> logits
    energy_kernel<<<(BB * SS) / 32, 32, 0, stream>>>(enc, weB, decproj, vw, logits);
    // K3: softmax over S
    softmax_kernel<<<BB, SS, 0, stream>>>(logits, mask, weights);
    // K4: weighted sum of enc rows
    wsum_kernel<<<BB * (EE / 256), 256, 0, stream>>>(enc, weights, out);
}
